// ConvShapeletModel_68702296867283
// MI455X (gfx1250) — matrix operand order repaired and checked by reading, not compiled, not run on hardware
//
#include <hip/hip_runtime.h>
#include <math.h>

typedef __attribute__((ext_vector_type(16))) __bf16 v16bf;
typedef __attribute__((ext_vector_type(8)))  float  v8f;

#define BB    128
#define DD    50
#define TT    512
#define HH    128
#define CC    12
#define NSHPC 384
#define TPAD  513   // conv output length T+1
#define KSH   12    // head K-steps: 384/32

// fp32 -> bf16 (round-half-up), scalar and packed-pair forms
__device__ __forceinline__ unsigned short f2bf16(float f) {
  union { float f; unsigned u; } c; c.f = f;
  return (unsigned short)((c.u + 0x8000u) >> 16);
}
__device__ __forceinline__ unsigned packbf(float f0, float f1) {
  union { float f; unsigned u; } a, b; a.f = f0; b.f = f1;
  return ((a.u + 0x8000u) >> 16) | ((b.u + 0x8000u) & 0xFFFF0000u);
}

union BfFrag { v16bf v; uint4 q[2]; unsigned u[8]; };

// fast sigmoid: v_rcp_f32 (~1 ulp) is exact enough for a bf16-bound result
__device__ __forceinline__ float sigf(float x) {
  return __builtin_amdgcn_rcpf(1.0f + __expf(-x));
}

// ---------------------------------------------------------------------------
// Pre-pack conv weights (H, D*S) into bf16 WMMA A-fragments, zero-padded to
// K multiple of 32.  Layout: ap[((ht*KS + ks)*32 + lane)*8 + v]  (dwords).
// A-fragment (16x32 bf16): lane m=L&15 is row; VGPR v holds K pair
//   k0 = 2v + 16*(L>=16) (contiguous order chosen here; B matches).
// ---------------------------------------------------------------------------
template<int S>
__global__ void prep_apack_kernel(const float* __restrict__ w,
                                  unsigned* __restrict__ ap) {
  constexpr int KTOT = DD * S;
  constexpr int KS   = (KTOT + 31) / 32;
  const int idx = blockIdx.x * blockDim.x + threadIdx.x;   // (ht*KS+ks)*32+lane
  if (idx >= 8 * KS * 32) return;
  const int lane = idx & 31;
  const int ks   = (idx >> 5) % KS;
  const int ht   = (idx >> 5) / KS;
  const int half = lane >> 4, mrow = lane & 15;
  const float* wr = w + (size_t)(ht * 16 + mrow) * KTOT + ks * 32;
  unsigned* o = ap + (size_t)idx * 8;
#pragma unroll
  for (int v = 0; v < 8; ++v) {
    const int k0 = 2 * v + 16 * half;
    const int kg = ks * 32 + k0;
    float f0 = 0.f, f1 = 0.f;
    if (kg < KTOT)     f0 = wr[k0];
    if (kg + 1 < KTOT) f1 = wr[k0 + 1];
    o[v] = packbf(f0, f1);
  }
}

// Pre-pack head weights: cols 0..11 = logreg rows, col 12 = dec_w, 13..15 = 0.
// B-fragment (32x16 bf16): lane n=L&15 is col; here VGPR j holds K pair 2j+16*(L>=16) (A matches).
__global__ void prep_hpack_kernel(const float* __restrict__ lw,
                                  const float* __restrict__ dw,
                                  unsigned* __restrict__ hp) {
  const int idx = blockIdx.x * blockDim.x + threadIdx.x;   // ks*32 + lane
  if (idx >= KSH * 32) return;
  const int lane = idx & 31, ks = idx >> 5;
  const int half = lane >> 4, col = lane & 15;
#pragma unroll
  for (int j = 0; j < 8; ++j) {
    float f[2];
#pragma unroll
    for (int e = 0; e < 2; ++e) {
      const int k = ks * 32 + 2 * j + 16 * half + e;
      float wv = 0.f;
      if (col < CC)       wv = lw[col * NSHPC + k];
      else if (col == CC) wv = dw[k];
      f[e] = wv;
    }
    hp[(size_t)idx * 8 + j] = packbf(f[0], f[1]);
  }
}

// ---------------------------------------------------------------------------
// Conv-as-GEMM with LDS-staged im2col.
// Block = 256 threads (8 waves), covers 32 time cols x all 128 H channels.
// x slab (<=52 x S+32 bf16) staged in LDS once; each wave does 2 h-tiles
// against one shared B fragment -> 2 WMMAs per K-step.
// ---------------------------------------------------------------------------
template<int S>
__global__ void __launch_bounds__(256)
conv_wmma_kernel(const float* __restrict__ x, const unsigned* __restrict__ apack,
                 const float* __restrict__ bias, float* __restrict__ raw) {
  constexpr int KTOT = DD * S;
  constexpr int KS   = (KTOT + 31) / 32;
  constexpr int WP   = S + 32;           // slab width: t0-S .. t0+31
  constexpr int NROW = 52;               // >= max (KS*32-1)/S + 1; rows>=DD are 0
  __shared__ unsigned short xs[NROW * WP];

  const int tid  = threadIdx.x;
  const int lane = tid & 31;
  const int wave = tid >> 5;
  const int half = lane >> 4;
  const int m    = lane & 15;
  const int t0   = blockIdx.x * 32;
  const int b    = blockIdx.y;

  // Cooperative slab fill: zero-padded (left conv padding, t>=T, d>=D), bf16.
  const float* xb = x + (size_t)b * DD * TT;
  for (int i = tid; i < NROW * WP; i += 256) {
    const int d = i / WP;
    const int tt = i - d * WP;
    const int time = t0 - S + tt;
    float v = 0.f;
    if (d < DD && time >= 0 && time < TT) v = xb[d * TT + time];
    xs[i] = f2bf16(v);
  }
  __syncthreads();

  const int ht0 = (wave & 3) * 2;        // this wave's pair of h-tiles
  const int tw  = (wave >> 2) * 16;      // which 16-col half of the slab
  const unsigned* ap0 = apack + ((size_t)(ht0 * KS) * 32 + lane) * 8;
  const unsigned* ap1 = apack + ((size_t)((ht0 + 1) * KS) * 32 + lane) * 8;

  v8f acc0 = {0.f,0.f,0.f,0.f,0.f,0.f,0.f,0.f};
  v8f acc1 = {0.f,0.f,0.f,0.f,0.f,0.f,0.f,0.f};

  for (int ks = 0; ks < KS; ++ks) {
    __builtin_prefetch(ap0 + 512, 0, 0);             // global_prefetch_b8
    BfFrag A0, A1, Bf;
    A0.q[0] = ((const uint4*)ap0)[0];  A0.q[1] = ((const uint4*)ap0)[1];
    A1.q[0] = ((const uint4*)ap1)[0];  A1.q[1] = ((const uint4*)ap1)[1];

    const int base = ks * 32 + 16 * half;
    const int coff = tw + m;
#pragma unroll
    for (int j = 0; j < 8; ++j) {
      const int k0 = base + 2 * j;
      const int d0 = k0 / S;                         // compile-time S: magic mul
      const int r0 = k0 - d0 * S;
      const int i0 = d0 * WP + r0 + coff;
      const int i1 = i0 + ((r0 == S - 1) ? (WP - S + 1) : 1);
      const unsigned lo = xs[i0];
      const unsigned hi = xs[i1];
      Bf.u[j] = lo | (hi << 16);
    }
    acc0 = __builtin_amdgcn_wmma_f32_16x16x32_bf16(false, A0.v, false, Bf.v,
                                                   (short)0, acc0, false, false);
    acc1 = __builtin_amdgcn_wmma_f32_16x16x32_bf16(false, A1.v, false, Bf.v,
                                                   (short)0, acc1, false, false);
    ap0 += 256; ap1 += 256;
  }

  const int t = t0 + tw + m;                         // conv output position
  if (t < TPAD) {
    const int h0 = ht0 * 16;
#pragma unroll
    for (int r = 0; r < 8; ++r) {
      const int mm = r + 8 * half;
      raw[(size_t)(b * HH + h0 + mm) * TPAD + t]      = acc0[r] + bias[h0 + mm];
      raw[(size_t)(b * HH + h0 + 16 + mm) * TPAD + t] = acc1[r] + bias[h0 + 16 + mm];
    }
  }
}

// ---------------------------------------------------------------------------
// Prefix-max over time per (b,channel); transpose to (B,T,NSHP); BN partials.
// ---------------------------------------------------------------------------
__global__ void cummax_stats_kernel(const float* __restrict__ raw,
                                    float* __restrict__ feat,
                                    float* __restrict__ ssum,
                                    float* __restrict__ ssq) {
  const int tid = blockIdx.x * blockDim.x + threadIdx.x;
  if (tid >= BB * NSHPC) return;
  const int b  = tid / NSHPC;
  const int ch = tid % NSHPC;
  const int g  = ch / HH;
  const int h  = ch % HH;
  const float* p = raw + (size_t)g * BB * HH * TPAD + (size_t)(b * HH + h) * TPAD;
  float run = -INFINITY, s = 0.f, sq = 0.f;
  for (int t = 0; t < TT; ++t) {               // only first T of T+1 are used
    run = fmaxf(run, p[t]);
    feat[(size_t)(b * TT + t) * NSHPC + ch] = run;
    s += run; sq += run * run;
  }
  atomicAdd(&ssum[ch], s);
  atomicAdd(&ssq[ch], sq);
}

__global__ void zero_kernel(float* __restrict__ p, int n) {
  const int i = blockIdx.x * blockDim.x + threadIdx.x;
  if (i < n) p[i] = 0.f;
}

__global__ void bn_finalize_kernel(const float* __restrict__ ssum,
                                   const float* __restrict__ ssq,
                                   const float* __restrict__ gamma,
                                   const float* __restrict__ beta,
                                   float* __restrict__ scale,
                                   float* __restrict__ shift) {
  const int c = threadIdx.x;
  if (c >= NSHPC) return;
  const float inv_n = 1.0f / (float)(BB * TT);
  const float mean = ssum[c] * inv_n;
  const float var  = ssq[c] * inv_n - mean * mean;   // biased var (BN train)
  const float sc   = rsqrtf(var + 1e-5f) * gamma[c];
  scale[c] = sc;
  shift[c] = beta[c] - mean * sc;
}

// ---------------------------------------------------------------------------
// Head: rows=(b,t), K=384, N=16 (0..11 logreg, 12 dec).  Two accumulators
// (A1=BN(feat), A2=sigmoid(BN(feat))) share one pre-packed B fragment.
// Epilogue: per-row log_softmax via LDS tile, and deltas.
// ---------------------------------------------------------------------------
__global__ void __launch_bounds__(32)
head_wmma_kernel(const float* __restrict__ feat,
                 const float* __restrict__ scale, const float* __restrict__ shift,
                 const unsigned* __restrict__ hpack,
                 const float* __restrict__ lb, const float* __restrict__ db,
                 float* __restrict__ out) {
  const int lane = threadIdx.x;
  const int half = lane >> 4;
  const int m    = lane & 15;
  const int row0 = blockIdx.x * 16;

  __shared__ float sscl[NSHPC];
  __shared__ float sshf[NSHPC];
  for (int i = lane; i < NSHPC; i += 32) { sscl[i] = scale[i]; sshf[i] = shift[i]; }
  __syncthreads();

  v8f c1 = {0.f,0.f,0.f,0.f,0.f,0.f,0.f,0.f};
  v8f c2 = {0.f,0.f,0.f,0.f,0.f,0.f,0.f,0.f};
  const float* frow = feat + (size_t)(row0 + m) * NSHPC;
  const unsigned* hp = hpack + (size_t)lane * 8;

  for (int ks = 0; ks < KSH; ++ks) {
    BfFrag A1, A2, Bf;
    Bf.q[0] = ((const uint4*)hp)[0];  Bf.q[1] = ((const uint4*)hp)[1];
#pragma unroll
    for (int v = 0; v < 8; ++v) {
      const int k0 = 2 * v + 16 * half;
      const int ka = ks * 32 + k0;
      const float fa = frow[ka]     * sscl[ka]     + sshf[ka];
      const float fb = frow[ka + 1] * sscl[ka + 1] + sshf[ka + 1];
      A1.u[v] = packbf(fa, fb);
      A2.u[v] = packbf(sigf(fa), sigf(fb));
    }
    c1 = __builtin_amdgcn_wmma_f32_16x16x32_bf16(false, A1.v, false, Bf.v,
                                                 (short)0, c1, false, false);
    c2 = __builtin_amdgcn_wmma_f32_16x16x32_bf16(false, A2.v, false, Bf.v,
                                                 (short)0, c2, false, false);
    hp += 256;
  }

  __shared__ float slog[16][17];
  __shared__ float sdec[16];
#pragma unroll
  for (int r = 0; r < 8; ++r) {
    const int mm = r + 8 * half;
    if (m < CC)  slog[mm][m] = c1[r] + lb[m];
    if (m == CC) sdec[mm]    = c2[r];
  }
  __syncthreads();

  if (lane < 16) {
    const int R = row0 + lane;
    float mx = -INFINITY;
#pragma unroll
    for (int n = 0; n < CC; ++n) mx = fmaxf(mx, slog[lane][n]);
    float se = 0.f;
#pragma unroll
    for (int n = 0; n < CC; ++n) se += __expf(slog[lane][n] - mx);
    const float ls = __logf(se) + mx;
#pragma unroll
    for (int n = 0; n < CC; ++n) out[(size_t)R * CC + n] = slog[lane][n] - ls;
    out[(size_t)BB * TT * CC + R] = sigf(sdec[lane] + db[0]);   // deltas
  }
}

// ---------------------------------------------------------------------------
// Attention-budget scan over time (sequential per batch).
// ---------------------------------------------------------------------------
__global__ void budget_kernel(float* __restrict__ out) {
  const int b = threadIdx.x;
  if (b >= BB) return;
  const float* deltas = out + (size_t)BB * TT * CC;
  float* pts = out + (size_t)BB * TT * CC + (size_t)BB * TT;
  float* bud = pts + (size_t)BB * TT;
  float bp = 1.0f;
  for (int t = 0; t < TT; ++t) {
    const float d = deltas[b * TT + t];
    const float p = (t == TT - 1) ? bp : d * bp;
    pts[b * TT + t] = p;
    bud[b * TT + t] = bp - p;
    bp *= (1.0f - d);
  }
}

extern "C" void kernel_launch(void* const* d_in, const int* in_sizes, int n_in,
                              void* d_out, int out_size, void* d_ws, size_t ws_size,
                              hipStream_t stream) {
  (void)in_sizes; (void)n_in; (void)out_size; (void)ws_size;
  const float* x   = (const float*)d_in[0];
  const float* w10 = (const float*)d_in[1];
  const float* b10 = (const float*)d_in[2];
  const float* w20 = (const float*)d_in[3];
  const float* b20 = (const float*)d_in[4];
  const float* w30 = (const float*)d_in[5];
  const float* b30 = (const float*)d_in[6];
  const float* gam = (const float*)d_in[7];
  const float* bet = (const float*)d_in[8];
  const float* lw  = (const float*)d_in[9];
  const float* lb  = (const float*)d_in[10];
  const float* dw  = (const float*)d_in[11];
  const float* db  = (const float*)d_in[12];
  float* out = (float*)d_out;

  // workspace layout
  float* ws = (float*)d_ws;
  const size_t RAW1 = (size_t)BB * HH * TPAD;            // per-size conv output
  float* raw   = ws;                                      // 3*RAW1
  float* feat  = ws + 3 * RAW1;                           // B*T*NSHP
  float* stats = feat + (size_t)BB * TT * NSHPC;          // 1536
  float* ssum  = stats;
  float* ssq   = stats + 384;
  float* scl   = stats + 768;
  float* shf   = stats + 1152;
  unsigned* ap10 = (unsigned*)(stats + 1536);             // 8*16*32*8 dwords
  unsigned* ap20 = ap10 + 8 * 16 * 32 * 8;                // 8*32*32*8
  unsigned* ap30 = ap20 + 8 * 32 * 32 * 8;                // 8*47*32*8
  unsigned* hp   = ap30 + 8 * 47 * 32 * 8;                // 12*32*8

  zero_kernel<<<(1536 + 255) / 256, 256, 0, stream>>>(stats, 1536);

  prep_apack_kernel<10><<<(8 * 16 * 32 + 255) / 256, 256, 0, stream>>>(w10, ap10);
  prep_apack_kernel<20><<<(8 * 32 * 32 + 255) / 256, 256, 0, stream>>>(w20, ap20);
  prep_apack_kernel<30><<<(8 * 47 * 32 + 255) / 256, 256, 0, stream>>>(w30, ap30);
  prep_hpack_kernel<<<2, 256, 0, stream>>>(lw, dw, hp);

  const dim3 cgrid((TPAD + 31) / 32, BB);                 // 17 x 128 blocks
  conv_wmma_kernel<10><<<cgrid, 256, 0, stream>>>(x, ap10, b10, raw);
  conv_wmma_kernel<20><<<cgrid, 256, 0, stream>>>(x, ap20, b20, raw + RAW1);
  conv_wmma_kernel<30><<<cgrid, 256, 0, stream>>>(x, ap30, b30, raw + 2 * RAW1);

  cummax_stats_kernel<<<(BB * NSHPC + 255) / 256, 256, 0, stream>>>(raw, feat, ssum, ssq);
  bn_finalize_kernel<<<1, NSHPC, 0, stream>>>(ssum, ssq, gam, bet, scl, shf);
  head_wmma_kernel<<<(BB * TT) / 16, 32, 0, stream>>>(feat, scl, shf, hp, lb, db, out);
  budget_kernel<<<1, BB, 0, stream>>>(out);
}
